// Criterion_8761733284571
// MI455X (gfx1250) — matrix operand order repaired and checked by reading, not compiled, not run on hardware
//
#include <hip/hip_runtime.h>
#include <hip/hip_bf16.h>

// ---------------------------------------------------------------------------
// Sinkhorn-OT pairwise criterion for MI455X (gfx1250, wave32, WMMA).
//
//   batch: [64, 128, 7, 7] f32, labels: [64] int
//
// Stage 1 (prep):   per image, L2-normalize the 128-dim channel vector at each
//                   of the 49 spatial sites + the normalized channel-mean, and
//                   pack into a bf16 tile F[n][64][128] (row = spatial site,
//                   row 49 = normalized mean, rows 50..63 = 0, col = channel).
// Stage 2 (pairs):  block (i,j): stage F_i via the Tensor Data Mover (one D#
//                   descriptor, TENSORcnt) and F_j via the per-lane async
//                   Global->LDS engine (ASYNCcnt) -- both CDNA5 DMA paths run
//                   concurrently, no VGPR round trip. Then sim[64][64] =
//                   F_i * F_j^T via v_wmma_f32_16x16x32_bf16, giving sim1
//                   (49x49), att_u (row 49), att_v (col 49) and sim2
//                   ([49][49]) in one GEMM. Then f32 Sinkhorn in LDS
//                   (<=100 iters, early stop on mean |r - r0| < 0.1) and the
//                   transport-weighted reduction -> simMat[i][j].
// Stage 3 (loss):   64-anchor margin/log-sum-exp loss -> scalar.
// ---------------------------------------------------------------------------

typedef __attribute__((ext_vector_type(16))) __bf16 v16bf;
typedef __attribute__((ext_vector_type(8)))  __bf16 v8bf;
typedef __attribute__((ext_vector_type(8)))  float  v8f;
typedef __attribute__((ext_vector_type(4)))  unsigned int u32x4;
typedef __attribute__((ext_vector_type(8)))  int          i32x8;
typedef __attribute__((ext_vector_type(4)))  int          i32x4;

#define NB      64      // batch
#define NC      128     // channels (K dim)
#define NS      49      // spatial sites
#define SP      64      // padded spatial (M = N = 64)
#define EPS_INV 20.0f   // 1 / 0.05
#define KPAD    52      // padded row stride of K matrix in LDS

// ---------------------------------------------------------------------------
// Kernel 1: normalize features, emit bf16 WMMA-friendly tiles.
// ---------------------------------------------------------------------------
__global__ __launch_bounds__(128) void prep_kernel(const float* __restrict__ batch,
                                                   __bf16* __restrict__ F) {
    const int n = blockIdx.x;
    const int t = threadIdx.x;

    __shared__ float f[NC * NS];       // f[c][s]
    __shared__ float invn[NS];
    __shared__ float meanv[NC];
    __shared__ float invMean;

    const float* g = batch + (size_t)n * NC * NS;
    for (int idx = t; idx < NC * NS; idx += 128) f[idx] = g[idx];
    __syncthreads();

    // per-channel spatial mean (raw, pre-normalization)
    float s = 0.0f;
    for (int j = 0; j < NS; ++j) s += f[t * NS + j];
    meanv[t] = s * (1.0f / (float)NS);

    // per-site channel-vector L2 norm
    if (t < NS) {
        float q = 0.0f;
        for (int c = 0; c < NC; ++c) { float x = f[c * NS + t]; q += x * x; }
        invn[t] = 1.0f / fmaxf(sqrtf(q), 1e-12f);
    }
    __syncthreads();

    if (t == 0) {
        float q = 0.0f;
        for (int c = 0; c < NC; ++c) q += meanv[c] * meanv[c];
        invMean = 1.0f / fmaxf(sqrtf(q), 1e-12f);
    }
    __syncthreads();

    __bf16* Fo = F + (size_t)n * SP * NC;   // [64 rows][128 cols], row-major
    for (int row = 0; row < SP; ++row) {
        float val;
        if (row < NS)       val = f[t * NS + row] * invn[row];
        else if (row == NS) val = meanv[t] * invMean;
        else                val = 0.0f;
        Fo[row * NC + t] = (__bf16)val;
    }
}

// ---------------------------------------------------------------------------
// Kernel 2: one block per (i,j) pair. 128 threads = 4 waves (wave32).
// ---------------------------------------------------------------------------
__global__ __launch_bounds__(128) void pair_kernel(const __bf16* __restrict__ F,
                                                   float* __restrict__ simMat) {
    const int pair = blockIdx.x;
    const int i = pair >> 6;
    const int j = pair & 63;
    const int t    = threadIdx.x;
    const int lane = t & 31;
    const int w    = t >> 5;

    __shared__ uint4 sFi4[1024];          // 16 KB: F_i tile (A, 64x128 bf16)
    __shared__ uint4 sFj4[1024];          // 16 KB: F_j tile (B^T, 64x128 bf16)
    __shared__ float sim[SP * SP];        // 16 KB: GEMM result
    __shared__ float Km[NS * KPAD];       // Gibbs kernel
    __shared__ float u[SP], v[SP], rv[SP], cv[SP], dif[SP];
    __shared__ float red[128];
    __shared__ float errS;

    // ---- CDNA5 DMA staging: TDM for F_i, async Global->LDS for F_j --------
    {
        const char* gbi = (const char*)(F + (size_t)i * SP * NC);
        const char* gbj = (const char*)(F + (size_t)j * SP * NC);
        const unsigned ldsI = (unsigned)(unsigned long long)(&sFi4[0]);
        const unsigned ldsJ = (unsigned)(unsigned long long)(&sFj4[0]);

#if __has_builtin(__builtin_amdgcn_tensor_load_to_lds)
        // F_i: single Tensor Data Mover descriptor (1-D tile, 4096 dwords).
        // D# group0: {count=1 | gather off, lds_addr, gaddr_lo,
        //             gaddr[56:32] | type=2<<30}
        // D# group1: data_size=4B (bits17:16=2); tensor_dim0=4096
        //            (d1[31:16]=0x1000); tensor_dim1=1 (d2[31:16]=1);
        //            tile_dim0=4096 (d3[31:16]=0x1000); tile_dim1=1 (d4=1);
        //            tensor_dim0_stride=4096 (d5); rest 0.
        // Wave-uniform issue (wave 0); TDM ignores EXEC, done once per op.
        if (t < 32) {
            const unsigned long long ga = (unsigned long long)gbi;
            u32x4 g0 = { 1u, ldsI, (unsigned)ga,
                         (unsigned)((ga >> 32) & 0x1FFFFFFull) | (2u << 30) };
            i32x8 g1 = { 0x20000, 0x10000000, 0x10000, 0x10000000,
                         1, 4096, 0, 0 };
            i32x4 z4 = { 0, 0, 0, 0 };
            i32x8 z8 = { 0, 0, 0, 0, 0, 0, 0, 0 };
            // 6-arg toolchain flavor: (g0, g1, g2, g3, extra, cpol)
            __builtin_amdgcn_tensor_load_to_lds(g0, g1, z4, z4, z8, 0);
            __builtin_amdgcn_s_wait_tensorcnt(0);
        }
#else
        __builtin_prefetch(gbi, 0, 3);
#pragma unroll
        for (int itc = 0; itc < 8; ++itc) {
            const unsigned off = (unsigned)(t + itc * 128) * 16u;
            asm volatile("global_load_async_to_lds_b128 %0, %1, %2"
                         :: "v"(ldsI + off), "v"(off), "s"(gbi) : "memory");
        }
#endif

        // F_j: per-lane async Global->LDS (ASYNCcnt engine), GVS mode:
        // vdst = per-lane LDS byte address, vaddr = 32-bit offset, saddr = base.
        __builtin_prefetch(gbj, 0, 3);
#pragma unroll
        for (int itc = 0; itc < 8; ++itc) {
            const unsigned off = (unsigned)(t + itc * 128) * 16u;
            asm volatile("global_load_async_to_lds_b128 %0, %1, %2"
                         :: "v"(ldsJ + off), "v"(off), "s"(gbj) : "memory");
        }
        asm volatile("s_wait_asynccnt 0x0" ::: "memory");
    }
    __syncthreads();

    const __bf16* A  = (const __bf16*)sFi4;
    const __bf16* Bt = (const __bf16*)sFj4;

    // --- WMMA GEMM: sim[64][64] = A(64x128) * Bt(64x128)^T ------------------
    // A lane layout (16-bit A 16x32): lanes 0-15 -> K {kb..kb+7, kb+16..kb+23},
// lanes 16-31 shift K by +8. B (32x16): same K mapping as A,
// column N = lane&15.
    v8f acc[4] = {};
    const int mrow  = (w << 4) + (lane & 15);
    const int ncol  = lane & 15;
    const int kselA = (lane >= 16) ? 8 : 0;
    const int kselB = (lane >= 16) ? 8 : 0;

    for (int kk = 0; kk < 4; ++kk) {
        const int kb = kk << 5;
        const __bf16* pa = A + mrow * NC + kb + kselA;
        v8bf alo = *(const v8bf*)(pa);
        v8bf ahi = *(const v8bf*)(pa + 16);
        v16bf av = __builtin_shufflevector(alo, ahi,
            0, 1, 2, 3, 4, 5, 6, 7, 8, 9, 10, 11, 12, 13, 14, 15);
#pragma unroll
        for (int nt = 0; nt < 4; ++nt) {
            const __bf16* pb = Bt + ((nt << 4) + ncol) * NC + kb + kselB;
            v8bf blo = *(const v8bf*)(pb);
            v8bf bhi = *(const v8bf*)(pb + 16);
            v16bf bv = __builtin_shufflevector(blo, bhi,
                0, 1, 2, 3, 4, 5, 6, 7, 8, 9, 10, 11, 12, 13, 14, 15);
            acc[nt] = __builtin_amdgcn_wmma_f32_16x16x32_bf16(
                false, av, false, bv, (short)0, acc[nt], false, false);
        }
    }

    // scatter accumulators: VGPR r -> M = 16w + r + (lane>=16 ? 8 : 0)
#pragma unroll
    for (int nt = 0; nt < 4; ++nt) {
#pragma unroll
        for (int r8 = 0; r8 < 8; ++r8) {
            const int M = (w << 4) + r8 + ((lane >= 16) ? 8 : 0);
            const int N = (nt << 4) + ncol;
            sim[M * SP + N] = acc[nt][r8];
        }
    }
    __syncthreads();

    const float sim2 = sim[NS * SP + NS];   // xmean . ymean

    // marginals u (from row 49 = att_u) and v (from col 49 = att_v)
    if (t < NS) {
        float su = 0.0f, sv = 0.0f;
        for (int m = 0; m < NS; ++m) {
            su += fmaxf(sim[NS * SP + m], 0.0f);
            sv += fmaxf(sim[m * SP + NS], 0.0f);
        }
        u[t]  = fmaxf(sim[NS * SP + t], 0.0f) / (su + 1e-5f);
        v[t]  = fmaxf(sim[t * SP + NS], 0.0f) / (sv + 1e-5f);
        rv[t] = 1.0f;
        cv[t] = 1.0f;
    }
    // Gibbs kernel K = exp((sim1 - 1) / eps)
    for (int idx = t; idx < NS * NS; idx += 128) {
        const int s = idx / NS;
        const int m = idx - s * NS;
        Km[s * KPAD + m] = __expf((sim[s * SP + m] - 1.0f) * EPS_INV);
    }
    __syncthreads();

    // --- Sinkhorn iterations (early stop: mean |r - r0| < 0.1) -------------
    for (int it = 0; it < 100; ++it) {
        if (t < NS) {
            const float r0 = rv[t];
            float s = 0.0f;
            const float* kr = &Km[t * KPAD];
            for (int m = 0; m < NS; ++m) s += kr[m] * cv[m];
            const float rn = u[t] / s;
            rv[t]  = rn;
            dif[t] = fabsf(rn - r0);
        }
        __syncthreads();
        if (t < NS) {
            float s = 0.0f;
            for (int sr = 0; sr < NS; ++sr) s += Km[sr * KPAD + t] * rv[sr];
            cv[t] = v[t] / s;
        }
        if (t == 0) {
            float e = 0.0f;
            for (int s = 0; s < NS; ++s) e += dif[s];
            errS = e * (1.0f / (float)NS);
        }
        __syncthreads();
        if (errS < 0.1f) break;   // uniform: all threads read same LDS value
    }

    // --- result = sum_{s,m} r_s c_m K_sm * 0.5 (sim1_sm + sim2) ------------
    float part = 0.0f;
    for (int idx = t; idx < NS * NS; idx += 128) {
        const int s = idx / NS;
        const int m = idx - s * NS;
        part += rv[s] * cv[m] * Km[s * KPAD + m]
              * 0.5f * (sim[s * SP + m] + sim2);
    }
    red[t] = part;
    __syncthreads();
    if (t == 0) {
        float acc2 = 0.0f;
        for (int q = 0; q < 128; ++q) acc2 += red[q];
        simMat[pair] = acc2;
    }
}

// ---------------------------------------------------------------------------
// Kernel 3: margin / soft-plus loss over the 64x64 similarity matrix.
// ---------------------------------------------------------------------------
__global__ __launch_bounds__(64) void loss_kernel(const float* __restrict__ simMat,
                                                  const int* __restrict__ labels,
                                                  float* __restrict__ out) {
    const int a = threadIdx.x;
    __shared__ int   lab[NB];
    __shared__ float per[NB];
    __shared__ int   val[NB];

    lab[a] = labels[a];
    __syncthreads();

    const float* row = simMat + a * NB;
    const int la = lab[a];
    const float INF = __builtin_inff();

    float minPos = INF, maxNeg = -INF;
    bool anyP = false, anyN = false;
    for (int jj = 0; jj < NB; ++jj) {
        const float s = row[jj];
        const bool same = (lab[jj] == la);
        if (same && jj != a) { anyP = true; minPos = fminf(minPos, s); }
        if (!same)           { anyN = true; maxNeg = fmaxf(maxNeg, s); }
    }

    float ps = 0.0f, ns = 0.0f;
    bool selP = false, selN = false;
    for (int jj = 0; jj < NB; ++jj) {
        const float s = row[jj];
        const bool same = (lab[jj] == la);
        if (!same && (s + 0.1f > minPos)) { selN = true; ns += __expf(40.0f * (s - 0.5f)); }
        if (same && jj != a && (s - 0.1f < maxNeg)) { selP = true; ps += __expf(-2.0f * (s - 0.5f)); }
    }

    const bool rvalid = anyP && anyN && selP && selN;
    per[a] = rvalid ? (log1pf(ps) * 0.5f + log1pf(ns) * 0.025f) : 0.0f;
    val[a] = rvalid ? 1 : 0;
    __syncthreads();

    if (a == 0) {
        float s = 0.0f; int n = 0;
        for (int q = 0; q < NB; ++q) { s += per[q]; n += val[q]; }
        out[0] = s / fmaxf((float)n, 1.0f);
    }
}

// ---------------------------------------------------------------------------
extern "C" void kernel_launch(void* const* d_in, const int* in_sizes, int n_in,
                              void* d_out, int out_size, void* d_ws, size_t ws_size,
                              hipStream_t stream) {
    (void)in_sizes; (void)n_in; (void)out_size; (void)ws_size;

    const float* batch  = (const float*)d_in[0];
    const int*   labels = (const int*)d_in[1];
    float*       out    = (float*)d_out;

    // workspace: F tiles (64 * 64*128 bf16 = 1 MB) then simMat (16 KB)
    __bf16* F      = (__bf16*)d_ws;
    float*  simMat = (float*)((char*)d_ws + (size_t)NB * SP * NC * sizeof(__bf16));

    prep_kernel<<<NB, 128, 0, stream>>>(batch, F);
    pair_kernel<<<NB * NB, 128, 0, stream>>>(F, simMat);
    loss_kernel<<<1, NB, 0, stream>>>(simMat, labels, out);
}
